// GPS_58171037057263
// MI455X (gfx1250) — matrix operand order repaired and checked by reading, not compiled, not run on hardware
//
#include <hip/hip_runtime.h>

#define H      128
#define NTOT   32768
#define NNODES 1024
#define NB     32
#define NHEADS 4
#define NEDGE  524288
#define LAYERS 3

typedef __attribute__((ext_vector_type(16))) _Float16 v16h;
typedef __attribute__((ext_vector_type(8)))  float    v8f;

// ---------------------------------------------------------------------------
// fp32 -> f16 conversion (one-time weight shadow copies)
// ---------------------------------------------------------------------------
__global__ __launch_bounds__(256) void cvt_f16_kernel(
    _Float16* __restrict__ o, const float* __restrict__ x, int n)
{
    int i = blockIdx.x * 256 + threadIdx.x;
    if (i < n) o[i] = (_Float16)x[i];
}

// ---------------------------------------------------------------------------
// GINE edge message + scatter-add:  aggr[dst] += relu(x[src] + edge_attr)
// ---------------------------------------------------------------------------
__global__ __launch_bounds__(256) void gine_scatter_kernel(
    const float* __restrict__ x, const float* __restrict__ ea,
    const int* __restrict__ src, const int* __restrict__ dst,
    float* __restrict__ aggr)
{
    unsigned i = blockIdx.x * 256u + threadIdx.x;
    unsigned e = i >> 7;          // /H
    unsigned c = i & 127u;        // %H
    float m = x[src[e] * H + c] + ea[i];
    m = fmaxf(m, 0.f);
    atomicAdd(&aggr[dst[e] * H + c], m);
}

// ---------------------------------------------------------------------------
// elementwise dst = a + b   (dst may alias a)
// ---------------------------------------------------------------------------
__global__ __launch_bounds__(256) void add_kernel(
    float* __restrict__ dstp, const float* __restrict__ a,
    const float* __restrict__ b, int n)
{
    int i = blockIdx.x * 256 + threadIdx.x;
    if (i < n) dstp[i] = a[i] + b[i];
}

// ---------------------------------------------------------------------------
// BatchNorm stats over 32768 rows; one block per channel.
// ---------------------------------------------------------------------------
__global__ __launch_bounds__(256) void bn_stats_kernel(
    const float* __restrict__ X, float* __restrict__ meanp, float* __restrict__ rstdp)
{
    int c = blockIdx.x;
    float s = 0.f, s2 = 0.f;
    for (int r = threadIdx.x; r < NTOT; r += 256) {
        float v = X[r * H + c];
        s += v; s2 += v * v;
    }
    __shared__ float sh[256], sh2[256];
    sh[threadIdx.x] = s; sh2[threadIdx.x] = s2;
    __syncthreads();
    for (int o = 128; o > 0; o >>= 1) {
        if (threadIdx.x < o) {
            sh[threadIdx.x]  += sh[threadIdx.x + o];
            sh2[threadIdx.x] += sh2[threadIdx.x + o];
        }
        __syncthreads();
    }
    if (threadIdx.x == 0) {
        float m = sh[0] * (1.f / NTOT);
        float v = sh2[0] * (1.f / NTOT) - m * m;
        meanp[c] = m;
        rstdp[c] = rsqrtf(v + 1e-5f);
    }
}

// fused residual variant: stats of (X + R) without materializing the sum
__global__ __launch_bounds__(256) void bn_stats_res_kernel(
    const float* __restrict__ X, const float* __restrict__ R,
    float* __restrict__ meanp, float* __restrict__ rstdp)
{
    int c = blockIdx.x;
    float s = 0.f, s2 = 0.f;
    for (int r = threadIdx.x; r < NTOT; r += 256) {
        float v = X[r * H + c] + R[r * H + c];
        s += v; s2 += v * v;
    }
    __shared__ float sh[256], sh2[256];
    sh[threadIdx.x] = s; sh2[threadIdx.x] = s2;
    __syncthreads();
    for (int o = 128; o > 0; o >>= 1) {
        if (threadIdx.x < o) {
            sh[threadIdx.x]  += sh[threadIdx.x + o];
            sh2[threadIdx.x] += sh2[threadIdx.x + o];
        }
        __syncthreads();
    }
    if (threadIdx.x == 0) {
        float m = sh[0] * (1.f / NTOT);
        float v = sh2[0] * (1.f / NTOT) - m * m;
        meanp[c] = m;
        rstdp[c] = rsqrtf(v + 1e-5f);
    }
}

// ---------------------------------------------------------------------------
// BatchNorm apply (+optional relu):  Y = (X [+R] - mean)*rstd*g + b
// ---------------------------------------------------------------------------
__global__ __launch_bounds__(256) void bn_apply_kernel(
    float* __restrict__ Y, const float* __restrict__ X,
    const float* __restrict__ meanp, const float* __restrict__ rstdp,
    const float* __restrict__ g, const float* __restrict__ bt, int relu)
{
    int i = blockIdx.x * 256 + threadIdx.x;
    int c = i & (H - 1);
    float v = (X[i] - meanp[c]) * rstdp[c] * g[c] + bt[c];
    if (relu) v = fmaxf(v, 0.f);
    Y[i] = v;
}

__global__ __launch_bounds__(256) void bn_apply_res_kernel(
    float* __restrict__ Y, const float* __restrict__ X, const float* __restrict__ R,
    const float* __restrict__ meanp, const float* __restrict__ rstdp,
    const float* __restrict__ g, const float* __restrict__ bt, int relu)
{
    int i = blockIdx.x * 256 + threadIdx.x;
    int c = i & (H - 1);
    float v = (X[i] + R[i] - meanp[c]) * rstdp[c] * g[c] + bt[c];
    if (relu) v = fmaxf(v, 0.f);
    Y[i] = v;
}

// ---------------------------------------------------------------------------
// WMMA GEMM:  C[M,Nout] = act( A[M,K] @ W16[Nout,K]^T + bias )
// block = 256 threads = 8 waves; block tile = 128 rows x 64 cols
// W tile staged global(f16) -> LDS via GLOBAL_LOAD_ASYNC_TO_LDS_B128
// (rows padded by 8 halves: 272B pitch -> bank stride 4 mod 64, 16B aligned)
// A tile staged fp32 -> f16 in per-wave LDS region.
// ---------------------------------------------------------------------------
__global__ __launch_bounds__(256) void gemm_kernel(
    const float* __restrict__ A, const _Float16* __restrict__ W,
    const float* __restrict__ bias, float* __restrict__ C,
    int K, int Nout, int relu)
{
    const int Kp = K + 8;                        // 16-byte row padding
    __shared__ __align__(16) _Float16 Wl[64 * 264];
    __shared__ __align__(16) _Float16 Al[8][16 * 34];

    const int tid  = threadIdx.x;
    const int wv   = tid >> 5;
    const int ln   = tid & 31;
    const int half = ln >> 4;
    const int l15  = ln & 15;
    const int row0 = blockIdx.x * 128 + wv * 16;
    const int col0 = blockIdx.y * 64;

    // async-stage weight tile: 64 rows x K halves, in 16B chunks
    {
        const int cpr = K >> 3;                  // chunks per row
        for (int i = tid; i < 64 * cpr; i += 256) {
            int j = i / cpr, c = i - j * cpr;
            unsigned ldsoff = (unsigned)(unsigned long long)&Wl[j * Kp + c * 8];
            unsigned long long gaddr =
                (unsigned long long)(W + (size_t)(col0 + j) * K + c * 8);
            asm volatile("global_load_async_to_lds_b128 %0, %1, off"
                         :: "v"(ldsoff), "v"(gaddr) : "memory");
        }
        asm volatile("s_wait_asynccnt 0x0" ::: "memory");
    }
    __syncthreads();

    v8f acc[4] = { {}, {}, {}, {} };

    for (int kt = 0; kt < K; kt += 32) {
        // stage this wave's 16x32 A tile (fp32 -> f16)
        #pragma unroll
        for (int i = 0; i < 16; ++i)
            Al[wv][i * 34 + ln] = (_Float16)A[(row0 + i) * K + kt + ln];
        if (kt + 32 < K)
            __builtin_prefetch(A + (size_t)(row0 + l15) * K + kt + 32 + half * 16, 0, 0);

        // A fragment (16-bit A layout: M = lane&15, K pairs split by lane half)
        v16h a;
        #pragma unroll
        for (int r = 0; r < 8; ++r) {
            int kb = (r >> 2) * 16 + half * 8 + (r & 3) * 2;
            a[2*r]   = Al[wv][l15 * 34 + kb];
            a[2*r+1] = Al[wv][l15 * 34 + kb + 1];
        }

        #pragma unroll
        for (int ct = 0; ct < 4; ++ct) {
            // B fragment: B[k][n] = W[col0+ct*16+n][k]; N = lane&15, K = (r>>2)*16+half*8+(r&3)*2
            v16h b;
            #pragma unroll
            for (int r = 0; r < 8; ++r) {
                int kk = kt + (r >> 2) * 16 + half * 8 + (r & 3) * 2;
                b[2*r]   = Wl[(ct * 16 + l15) * Kp + kk];
                b[2*r+1] = Wl[(ct * 16 + l15) * Kp + kk + 1];
            }
            acc[ct] = __builtin_amdgcn_wmma_f32_16x16x32_f16(
                false, a, false, b, (short)0, acc[ct], false, false);
        }
    }

    // epilogue: D layout M = r + 8*half, N = lane&15
    #pragma unroll
    for (int ct = 0; ct < 4; ++ct) {
        #pragma unroll
        for (int r = 0; r < 8; ++r) {
            int row = row0 + r + half * 8;
            int col = col0 + ct * 16 + l15;
            float v = acc[ct][r] + bias[col];
            if (relu) v = fmaxf(v, 0.f);
            C[row * Nout + col] = v;
        }
    }
}

// ---------------------------------------------------------------------------
// Flash attention over one (graph, head): N=1024 keys streamed in tiles of 64.
// block = 256 threads = 8 waves; each wave owns 16 query rows (128 q / block).
// grid = B * NH * (N/128) = 1024 blocks.
// ---------------------------------------------------------------------------
__global__ __launch_bounds__(256) void attn_kernel(
    const float* __restrict__ qkv, float* __restrict__ outp)
{
    __shared__ _Float16 Kt[64 * 34];       // [key][dh]   (padded)
    __shared__ _Float16 Vt[32 * 66];       // [dh][key]   (transposed, padded)
    __shared__ _Float16 Pl[8][16 * 66];    // per-wave P tile [qrow][key]

    const int tid  = threadIdx.x;
    const int wv   = tid >> 5;
    const int ln   = tid & 31;
    const int half = ln >> 4;
    const int l15  = ln & 15;

    const int qb = blockIdx.x & 7;
    const int hh = (blockIdx.x >> 3) & 3;
    const int bb = blockIdx.x >> 5;
    const int q0 = qb * 128 + wv * 16;
    const float scale = 0.17677669529663687f;   // 1/sqrt(32)

    // Q fragment (A layout), loaded once, kept in registers
    v16h qa;
    {
        const float* qp = qkv + (size_t)(bb * NNODES + q0 + l15) * 384 + hh * 32;
        #pragma unroll
        for (int r = 0; r < 8; ++r) {
            int kb = (r >> 2) * 16 + half * 8 + (r & 3) * 2;
            qa[2*r]   = (_Float16)qp[kb];
            qa[2*r+1] = (_Float16)qp[kb + 1];
        }
    }

    v8f o0 = {}, o1 = {};
    float mrow[8], lrow[8];
    #pragma unroll
    for (int r = 0; r < 8; ++r) { mrow[r] = -3.0e38f; lrow[r] = 0.f; }

    for (int kt = 0; kt < 16; ++kt) {
        __syncthreads();
        const int kk0 = kt * 64;
        for (int i = tid; i < 64 * 32; i += 256) {
            int key = i >> 5, d = i & 31;
            const float* kp = qkv + (size_t)(bb * NNODES + kk0 + key) * 384 + 128 + hh * 32;
            Kt[key * 34 + d] = (_Float16)kp[d];
            Vt[d * 66 + key] = (_Float16)kp[128 + d];   // v = k + 128
        }
        __syncthreads();

        // scores: S[q,key] = Q @ K^T (K-dim = DH = 32, one WMMA per 16-key tile)
        v8f s[4];
        #pragma unroll
        for (int ct = 0; ct < 4; ++ct) {
            v16h kb;
            #pragma unroll
            for (int r = 0; r < 8; ++r) {
                int d = (r >> 2) * 16 + half * 8 + (r & 3) * 2;
                kb[2*r]   = Kt[(ct * 16 + l15) * 34 + d];
                kb[2*r+1] = Kt[(ct * 16 + l15) * 34 + d + 1];
            }
            v8f z = {};
            s[ct] = __builtin_amdgcn_wmma_f32_16x16x32_f16(
                false, qa, false, kb, (short)0, z, false, false);
        }

        // online softmax (row reductions across the 16-lane half)
        #pragma unroll
        for (int r = 0; r < 8; ++r) {
            float mx = -3.0e38f;
            #pragma unroll
            for (int ct = 0; ct < 4; ++ct) { s[ct][r] *= scale; mx = fmaxf(mx, s[ct][r]); }
            #pragma unroll
            for (int off = 1; off < 16; off <<= 1) mx = fmaxf(mx, __shfl_xor(mx, off));
            float mnew = fmaxf(mrow[r], mx);
            float corr = __expf(mrow[r] - mnew);
            float rs = 0.f;
            #pragma unroll
            for (int ct = 0; ct < 4; ++ct) {
                float p = __expf(s[ct][r] - mnew);
                s[ct][r] = p;
                rs += p;
            }
            #pragma unroll
            for (int off = 1; off < 16; off <<= 1) rs += __shfl_xor(rs, off);
            lrow[r] = lrow[r] * corr + rs;
            mrow[r] = mnew;
            o0[r] *= corr; o1[r] *= corr;
            #pragma unroll
            for (int ct = 0; ct < 4; ++ct)
                Pl[wv][(r + 8 * half) * 66 + ct * 16 + l15] = (_Float16)s[ct][r];
        }

        // O += P @ V   (two k-steps of 32 keys; two 16-wide dh tiles)
        #pragma unroll
        for (int ks = 0; ks < 2; ++ks) {
            v16h pa;
            #pragma unroll
            for (int r = 0; r < 8; ++r) {
                int kb2 = (r >> 2) * 16 + half * 8 + (r & 3) * 2;
                pa[2*r]   = Pl[wv][l15 * 66 + ks * 32 + kb2];
                pa[2*r+1] = Pl[wv][l15 * 66 + ks * 32 + kb2 + 1];
            }
            v16h vb0, vb1;
            #pragma unroll
            for (int r = 0; r < 8; ++r) {
                int key = ks * 32 + (r >> 2) * 16 + half * 8 + (r & 3) * 2;
                vb0[2*r]   = Vt[l15 * 66 + key];
                vb0[2*r+1] = Vt[l15 * 66 + key + 1];
                vb1[2*r]   = Vt[(16 + l15) * 66 + key];
                vb1[2*r+1] = Vt[(16 + l15) * 66 + key + 1];
            }
            o0 = __builtin_amdgcn_wmma_f32_16x16x32_f16(false, pa, false, vb0, (short)0, o0, false, false);
            o1 = __builtin_amdgcn_wmma_f32_16x16x32_f16(false, pa, false, vb1, (short)0, o1, false, false);
        }
    }

    #pragma unroll
    for (int r = 0; r < 8; ++r) {
        int node = bb * NNODES + q0 + r + 8 * half;
        float inv = 1.f / lrow[r];
        outp[node * H + hh * 32 + l15]      = o0[r] * inv;
        outp[node * H + hh * 32 + 16 + l15] = o1[r] * inv;
    }
}

// ---------------------------------------------------------------------------
// host orchestration
// ---------------------------------------------------------------------------
extern "C" void kernel_launch(void* const* d_in, const int* in_sizes, int n_in,
                              void* d_out, int out_size, void* d_ws, size_t ws_size,
                              hipStream_t stream) {
    (void)in_sizes; (void)n_in; (void)out_size; (void)ws_size;

    const float* x_in      = (const float*)d_in[0];
    const float* edge_attr = (const float*)d_in[1];
    const float* gine_w1   = (const float*)d_in[2];
    const float* gine_b1   = (const float*)d_in[3];
    const float* gine_bn_g = (const float*)d_in[4];
    const float* gine_bn_b = (const float*)d_in[5];
    const float* gine_w2   = (const float*)d_in[6];
    const float* gine_b2   = (const float*)d_in[7];
    const float* attn_wqkv = (const float*)d_in[8];
    const float* attn_bqkv = (const float*)d_in[9];
    const float* attn_wo   = (const float*)d_in[10];
    const float* attn_bo   = (const float*)d_in[11];
    const float* norm1_g   = (const float*)d_in[12];
    const float* norm1_b   = (const float*)d_in[13];
    const float* norm2_g   = (const float*)d_in[14];
    const float* norm2_b   = (const float*)d_in[15];
    const float* norm3_g   = (const float*)d_in[16];
    const float* norm3_b   = (const float*)d_in[17];
    const float* mlp_w1    = (const float*)d_in[18];
    const float* mlp_b1    = (const float*)d_in[19];
    const float* mlp_w2    = (const float*)d_in[20];
    const float* mlp_b2    = (const float*)d_in[21];
    const int*   edge_idx  = (const int*)d_in[22];
    const int*   srcp = edge_idx;
    const int*   dstp = edge_idx + NEDGE;

    float* ws = (float*)d_ws;
    const size_t MH = (size_t)NTOT * H;
    float* Au = ws;                 // current x
    float* Bu = Au + MH;            // aggr / h2 / h_local / ff2
    float* Cu = Bu + MH;            // h1 / o_proj / h_attn
    float* Du = Cu + MH;            // qkv [NTOT,384] / ff1 [NTOT,256]
    float* Eu = Du + (size_t)NTOT * 384;   // attn_o / combine "out"
    float* meanb = Eu + MH;
    float* rstdb = meanb + H;

    // f16 weight shadow buffers (all layers, contiguous per tensor)
    _Float16* w16 = (_Float16*)(rstdb + H);
    const int n_w1   = LAYERS * H * H;
    const int n_w2   = LAYERS * H * H;
    const int n_wqkv = LAYERS * 3 * H * H;
    const int n_wo   = LAYERS * H * H;
    const int n_mw1  = LAYERS * 2 * H * H;
    const int n_mw2  = LAYERS * 2 * H * H;
    _Float16* w1_16   = w16;
    _Float16* w2_16   = w1_16   + n_w1;
    _Float16* wqkv_16 = w2_16   + n_w2;
    _Float16* wo_16   = wqkv_16 + n_wqkv;
    _Float16* mw1_16  = wo_16   + n_wo;
    _Float16* mw2_16  = mw1_16  + n_mw1;

    const dim3 blk(256);
    cvt_f16_kernel<<<(n_w1 + 255) / 256,   blk, 0, stream>>>(w1_16,   gine_w1,   n_w1);
    cvt_f16_kernel<<<(n_w2 + 255) / 256,   blk, 0, stream>>>(w2_16,   gine_w2,   n_w2);
    cvt_f16_kernel<<<(n_wqkv + 255) / 256, blk, 0, stream>>>(wqkv_16, attn_wqkv, n_wqkv);
    cvt_f16_kernel<<<(n_wo + 255) / 256,   blk, 0, stream>>>(wo_16,   attn_wo,   n_wo);
    cvt_f16_kernel<<<(n_mw1 + 255) / 256,  blk, 0, stream>>>(mw1_16,  mlp_w1,    n_mw1);
    cvt_f16_kernel<<<(n_mw2 + 255) / 256,  blk, 0, stream>>>(mw2_16,  mlp_w2,    n_mw2);

    hipMemcpyAsync(Au, x_in, sizeof(float) * MH, hipMemcpyDeviceToDevice, stream);

    const int nElem = NTOT * H;
    const int gElem = nElem / 256;
    const dim3 gemmHH(NTOT / 128, H / 64);

    for (int i = 0; i < LAYERS; ++i) {
        const _Float16* w1   = w1_16   + (size_t)i * H * H;
        const _Float16* w2   = w2_16   + (size_t)i * H * H;
        const _Float16* wqkv = wqkv_16 + (size_t)i * 3 * H * H;
        const _Float16* wo   = wo_16   + (size_t)i * H * H;
        const _Float16* mw1  = mw1_16  + (size_t)i * 2 * H * H;
        const _Float16* mw2  = mw2_16  + (size_t)i * 2 * H * H;
        const float* b1   = gine_b1 + (size_t)i * H;
        const float* bng  = gine_bn_g + (size_t)i * H;
        const float* bnb  = gine_bn_b + (size_t)i * H;
        const float* b2   = gine_b2 + (size_t)i * H;
        const float* bqkv = attn_bqkv + (size_t)i * 3 * H;
        const float* bo   = attn_bo + (size_t)i * H;
        const float* n1g  = norm1_g + (size_t)i * H;
        const float* n1b  = norm1_b + (size_t)i * H;
        const float* n2g  = norm2_g + (size_t)i * H;
        const float* n2b  = norm2_b + (size_t)i * H;
        const float* n3g  = norm3_g + (size_t)i * H;
        const float* n3b  = norm3_b + (size_t)i * H;
        const float* mb1  = mlp_b1 + (size_t)i * 2 * H;
        const float* mb2  = mlp_b2 + (size_t)i * H;

        // ---- GINEConv
        hipMemsetAsync(Bu, 0, sizeof(float) * MH, stream);
        gine_scatter_kernel<<<(NEDGE * H) / 256, blk, 0, stream>>>(Au, edge_attr, srcp, dstp, Bu);
        add_kernel<<<gElem, blk, 0, stream>>>(Bu, Bu, Au, nElem);                 // x + aggr
        gemm_kernel<<<gemmHH, blk, 0, stream>>>(Bu, w1, b1, Cu, H, H, 0);         // h1
        bn_stats_kernel<<<H, blk, 0, stream>>>(Cu, meanb, rstdb);
        bn_apply_kernel<<<gElem, blk, 0, stream>>>(Cu, Cu, meanb, rstdb, bng, bnb, 1);
        gemm_kernel<<<gemmHH, blk, 0, stream>>>(Cu, w2, b2, Bu, H, H, 0);         // h2
        bn_stats_res_kernel<<<H, blk, 0, stream>>>(Bu, Au, meanb, rstdb);         // stats(h2+x)
        bn_apply_res_kernel<<<gElem, blk, 0, stream>>>(Bu, Bu, Au, meanb, rstdb, n1g, n1b, 0); // h_local

        // ---- global attention
        gemm_kernel<<<dim3(NTOT / 128, 384 / 64), blk, 0, stream>>>(Au, wqkv, bqkv, Du, H, 384, 0);
        attn_kernel<<<NB * NHEADS * (NNODES / 128), blk, 0, stream>>>(Du, Eu);
        gemm_kernel<<<gemmHH, blk, 0, stream>>>(Eu, wo, bo, Cu, H, H, 0);         // o @ wo^T
        bn_stats_res_kernel<<<H, blk, 0, stream>>>(Cu, Au, meanb, rstdb);         // stats(o+x)
        bn_apply_res_kernel<<<gElem, blk, 0, stream>>>(Cu, Cu, Au, meanb, rstdb, n2g, n2b, 0); // h_attn

        // ---- combine + FFN + norm3
        add_kernel<<<gElem, blk, 0, stream>>>(Eu, Bu, Cu, nElem);                 // out = h_local + h_attn
        gemm_kernel<<<dim3(NTOT / 128, 256 / 64), blk, 0, stream>>>(Eu, mw1, mb1, Du, H, 256, 1);
        gemm_kernel<<<gemmHH, blk, 0, stream>>>(Du, mw2, mb2, Bu, 256, H, 0);     // ff2
        bn_stats_res_kernel<<<H, blk, 0, stream>>>(Bu, Eu, meanb, rstdb);         // stats(out+ff)
        float* tgt = (i == LAYERS - 1) ? (float*)d_out : Au;
        bn_apply_res_kernel<<<gElem, blk, 0, stream>>>(tgt, Bu, Eu, meanb, rstdb, n3g, n3b,
                                                       (i < LAYERS - 1) ? 1 : 0);
    }
}
